// ScaleSelfAttention_52029233824264
// MI455X (gfx1250) — matrix operand order repaired and checked by reading, not compiled, not run on hardware
//
#include <hip/hip_runtime.h>

// ---------------------------------------------------------------------------
// CDNA5 (gfx1250, wave32) fused self-attention:
//   f = x@Wf, g = x@Wg, h = x@Wh  (1x1 convs == GEMMs)
//   out = x + gamma * softmax(f g^T) h
// All matmuls: V_WMMA_F32_16X16X32_F16 (f16 operands, f32 accum).
// Flash-attention: scores never hit HBM.  V tiles staged per-workgroup into
// LDS with GLOBAL_LOAD_ASYNC_TO_LDS_B128 (ASYNCcnt), double-buffered.
// ---------------------------------------------------------------------------

typedef __attribute__((ext_vector_type(16))) _Float16 v16h;
typedef __attribute__((ext_vector_type(8)))  _Float16 v8h;
typedef __attribute__((ext_vector_type(8)))  float    v8f;
typedef __attribute__((ext_vector_type(4)))  int      v4i;

#define WMMA16x16x32(A, B, C) \
  __builtin_amdgcn_wmma_f32_16x16x32_f16(false, (A), false, (B), (short)0, (C), false, false)

#define AS1 __attribute__((address_space(1)))
#define AS3 __attribute__((address_space(3)))

constexpr int kBatch = 4;
constexpr int kN     = 4096;          // tokens per batch
constexpr int kC     = 256;          // channels
constexpr int kD     = 32;           // head dim == one WMMA K
constexpr int kRows  = kBatch * kN;  // 16384

// One async 16-byte transfer per lane: global -> LDS, tracked by ASYNCcnt.
__device__ __forceinline__ void async_copy_b128(const _Float16* gsrc, _Float16* ldst) {
    __builtin_amdgcn_global_load_async_to_lds_b128((AS1 v4i*)gsrc, (AS3 v4i*)ldst, 0, 0);
}

// A-fragment element e -> K = kh*16 + e : two contiguous 8-chunks.
__device__ __forceinline__ v16h load_a_frag_f32(const float* row, int kh) {
    v8f a0 = *(const v8f*)(row + kh * 16);
    v8f a1 = *(const v8f*)(row + kh * 16 + 8);
    v16h A;
    #pragma unroll
    for (int j = 0; j < 8; ++j) { A[j] = (_Float16)a0[j]; A[8 + j] = (_Float16)a1[j]; }
    return A;
}

// ---------------------------------------------------------------------------
// Stage 1a: f and g projections (COUT=32 compile-time -> immediate-offset
// weight loads in one clause).  Waves 0-1: f cols 0-31; waves 2-3: g.
// ---------------------------------------------------------------------------
__global__ __launch_bounds__(128)
void proj_fg_kernel(const float* __restrict__ x,
                    const float* __restrict__ wf,
                    const float* __restrict__ wg,
                    _Float16* __restrict__ qo,
                    _Float16* __restrict__ ko)
{
    const int lane = threadIdx.x & 31;
    const int wave = threadIdx.x >> 5;
    const int n    = lane & 15;
    const int kh   = lane >> 4;
    const int row0 = blockIdx.x * 16;

    const float* W      = (wave < 2) ? wf : wg;
    _Float16*    dst    = (wave < 2) ? qo : ko;
    const int    col    = (wave & 1) * 16 + n;

    v8f acc = 0.0f;
    const float* xrow = x + (size_t)(row0 + n) * kC;

    #pragma unroll
    for (int kk = 0; kk < kC / 32; ++kk) {
        v16h A = load_a_frag_f32(xrow + kk * 32, kh);
        const float* wp = W + (size_t)(kk * 32 + kh * 16) * 32 + col;  // stride const
        v16h Bm;
        #pragma unroll
        for (int e = 0; e < 16; ++e) Bm[e] = (_Float16)wp[e * 32];
        acc = WMMA16x16x32(A, Bm, acc);
    }

    #pragma unroll
    for (int r = 0; r < 8; ++r)
        dst[(size_t)(row0 + r + kh * 8) * kD + col] = (_Float16)acc[r];
}

// ---------------------------------------------------------------------------
// Stage 1b: h projection, stored TRANSPOSED [C][token] f16 so attention V
// B-fragments are contiguous along keys.  COUT=256 compile-time.
// ---------------------------------------------------------------------------
__global__ __launch_bounds__(128)
void proj_h_kernel(const float* __restrict__ x,
                   const float* __restrict__ wh,
                   _Float16* __restrict__ vt)
{
    const int lane = threadIdx.x & 31;
    const int wave = threadIdx.x >> 5;
    const int n    = lane & 15;
    const int kh   = lane >> 4;
    const int row0 = blockIdx.x * 16;
    const int ct   = blockIdx.y * 4 + wave;      // 0..15 channel tile
    const int col  = ct * 16 + n;

    v8f acc = 0.0f;
    const float* xrow = x + (size_t)(row0 + n) * kC;

    #pragma unroll
    for (int kk = 0; kk < kC / 32; ++kk) {
        v16h A = load_a_frag_f32(xrow + kk * 32, kh);
        const float* wp = wh + (size_t)(kk * 32 + kh * 16) * kC + col;  // stride const
        v16h Bm;
        #pragma unroll
        for (int e = 0; e < 16; ++e) Bm[e] = (_Float16)wp[e * kC];
        acc = WMMA16x16x32(A, Bm, acc);
    }

    // rows r+8*kh are consecutive tokens -> one packed 16B store
    const int b    = row0 >> 12;
    const int tok0 = (row0 & (kN - 1)) + kh * 8;
    v8h pk;
    #pragma unroll
    for (int r = 0; r < 8; ++r) pk[r] = (_Float16)acc[r];
    *(v8h*)(vt + ((size_t)b * kC + col) * kN + tok0) = pk;
}

// ---------------------------------------------------------------------------
// Stage 2: flash attention + residual.  4 waves/block, one 16-query tile per
// wave, all same batch (4 | 256) -> V tiles shared through LDS, async
// double-buffered.  Each wave owns full C=256 output (16 f32 accum tiles).
// ---------------------------------------------------------------------------
__global__ __launch_bounds__(128)
void attn_kernel(const float* __restrict__ x,
                 const _Float16* __restrict__ qm,
                 const _Float16* __restrict__ km,
                 const _Float16* __restrict__ vt,
                 const float* __restrict__ gptr,
                 float* __restrict__ out)
{
    __shared__ _Float16 vls[2][kC][32];    // 2 x 16KB V tiles (channel-major)
    __shared__ _Float16 pbuf[4][16][32];   // per-wave P transpose scratch

    const int tid  = threadIdx.x;
    const int lane = tid & 31;
    const int wave = tid >> 5;
    const int n    = lane & 15;
    const int kh   = lane >> 4;

    const int qt   = blockIdx.x * 4 + wave;   // 0..1023 query tile
    const int b    = qt >> 8;                 // same for all 4 waves in block
    const int row0 = qt * 16;

    const float gamma = gptr[0];
    const _Float16* kb = km + (size_t)b * kN * kD;
    const _Float16* vb = vt + (size_t)b * kC * kN;

    // Q A-fragment, held for the whole loop
    v16h Q;
    {
        const _Float16* qrow = qm + (size_t)(row0 + n) * kD;
        v8h q0 = *(const v8h*)(qrow + kh * 16);
        v8h q1 = *(const v8h*)(qrow + kh * 16 + 8);
        #pragma unroll
        for (int j = 0; j < 8; ++j) { Q[j] = q0[j]; Q[8 + j] = q1[j]; }
    }

    float mrow[8], lrow[8];
    #pragma unroll
    for (int r = 0; r < 8; ++r) { mrow[r] = -3.0e38f; lrow[r] = 0.0f; }
    v8f o[16];
    #pragma unroll
    for (int t = 0; t < 16; ++t) o[t] = 0.0f;

    // issue one V tile (16KB = 1024 lane-chunks of 16B; 8 wave-ops per wave)
    auto issue_v = [&](int buf, int key0) {
        #pragma unroll
        for (int i = 0; i < 8; ++i) {
            const int linear = i * 128 + tid;
            const int ch  = linear >> 2;        // 0..255
            const int sub = linear & 3;         // 16B sub-chunk of 64B row
            async_copy_b128(vb + (size_t)ch * kN + key0 + sub * 8,
                            &vls[buf][ch][sub * 8]);
        }
    };

    issue_v(0, 0);

    for (int kt = 0; kt < kN / 32; ++kt) {
        const int cur  = kt & 1;
        const int key0 = kt * 32;

        if (kt + 1 < kN / 32) {
            issue_v(cur ^ 1, key0 + 32);               // prefetch next tile
            __builtin_amdgcn_s_wait_asynccnt(8);       // oldest batch done
        } else {
            __builtin_amdgcn_s_wait_asynccnt(0);
        }
        __syncthreads();                               // tile visible to all

        // --- QK: K B-fragments straight from global (small, contiguous) ---
        v16h K0 = *(const v16h*)(kb + (size_t)(key0 + n) * kD + kh * 16);
        v16h K1 = *(const v16h*)(kb + (size_t)(key0 + 16 + n) * kD + kh * 16);
        v8f z = 0.0f;
        v8f s0 = WMMA16x16x32(Q, K0, z);
        v8f s1 = WMMA16x16x32(Q, K1, z);

        // --- online softmax: row r+8*kh lives across 16 lanes of a half ---
        float sc[8];
        #pragma unroll
        for (int r = 0; r < 8; ++r) {
            float t = fmaxf(s0[r], s1[r]);
            t = fmaxf(t, __shfl_xor(t, 1));
            t = fmaxf(t, __shfl_xor(t, 2));
            t = fmaxf(t, __shfl_xor(t, 4));
            t = fmaxf(t, __shfl_xor(t, 8));
            const float mn  = fmaxf(mrow[r], t);
            const float scl = __expf(mrow[r] - mn);
            const float p0  = __expf(s0[r] - mn);
            const float p1  = __expf(s1[r] - mn);
            float rs = p0 + p1;
            rs += __shfl_xor(rs, 1);
            rs += __shfl_xor(rs, 2);
            rs += __shfl_xor(rs, 4);
            rs += __shfl_xor(rs, 8);
            lrow[r] = lrow[r] * scl + rs;
            mrow[r] = mn;
            sc[r]   = scl;
            pbuf[wave][r + kh * 8][n]      = (_Float16)p0;   // C->A transpose
            pbuf[wave][r + kh * 8][16 + n] = (_Float16)p1;
        }

        // P A-fragment reload (same-wave DS ops are in-order)
        v16h P;
        {
            const _Float16* pr = &pbuf[wave][n][kh * 16];
            v8h p0 = *(const v8h*)pr;
            v8h p1 = *(const v8h*)(pr + 8);
            #pragma unroll
            for (int j = 0; j < 8; ++j) { P[j] = p0[j]; P[8 + j] = p1[j]; }
        }

        // --- PV: V B-frags from shared LDS tile (contiguous 32B ds loads) ---
        #pragma unroll
        for (int t = 0; t < 16; ++t) {
            #pragma unroll
            for (int r = 0; r < 8; ++r) o[t][r] *= sc[r];
            v16h V = *(const v16h*)&vls[cur][t * 16 + n][kh * 16];
            o[t] = WMMA16x16x32(P, V, o[t]);
        }
        __syncthreads();   // all waves done reading before buffer reuse
    }

    // --- finalize: o/l, residual + gamma ---
    float inv[8];
    #pragma unroll
    for (int r = 0; r < 8; ++r) inv[r] = 1.0f / lrow[r];
    #pragma unroll
    for (int t = 0; t < 16; ++t) {
        const int c = t * 16 + n;
        #pragma unroll
        for (int r = 0; r < 8; ++r) {
            const size_t idx = (size_t)(row0 + r + kh * 8) * kC + c;
            out[idx] = x[idx] + gamma * (o[t][r] * inv[r]);
        }
    }
}

// ---------------------------------------------------------------------------
// Workspace (10 MB): [0,1MB) f | [1,2MB) g | [2,10MB) h^T [4][256][4096] f16
// ---------------------------------------------------------------------------
extern "C" void kernel_launch(void* const* d_in, const int* in_sizes, int n_in,
                              void* d_out, int out_size, void* d_ws, size_t ws_size,
                              hipStream_t stream) {
    (void)in_sizes; (void)n_in; (void)out_size; (void)ws_size;
    const float* x     = (const float*)d_in[0];
    const float* wf    = (const float*)d_in[1];
    const float* wg    = (const float*)d_in[2];
    const float* wh    = (const float*)d_in[3];
    const float* gamma = (const float*)d_in[4];

    char* ws = (char*)d_ws;
    _Float16* q  = (_Float16*)(ws);
    _Float16* k  = (_Float16*)(ws + ((size_t)1 << 20));
    _Float16* vt = (_Float16*)(ws + ((size_t)2 << 20));

    proj_fg_kernel<<<kRows / 16, 128, 0, stream>>>(x, wf, wg, q, k);

    dim3 gh(kRows / 16, 4);     // 1024 row tiles x 16 channel tiles
    proj_h_kernel<<<gh, 128, 0, stream>>>(x, wh, vt);

    attn_kernel<<<kRows / 16 / 4, 128, 0, stream>>>(x, q, k, vt, gamma, (float*)d_out);
}
